// SelfAttention_23639499997333
// MI455X (gfx1250) — matrix operand order repaired and checked by reading, not compiled, not run on hardware
//
#include <hip/hip_runtime.h>
#include <stdint.h>

// ---------------- types ----------------
typedef __attribute__((ext_vector_type(16))) __bf16 v16bf;
typedef __attribute__((ext_vector_type(16))) short  v16s;
typedef __attribute__((ext_vector_type(8)))  float  v8f;
typedef __attribute__((__vector_size__(16))) int    v4i;   // matches async-LDS builtin param

#define EMBED 1024
#define NHEAD 16
#define HDIM  64
#define SEQ   2048
#define BATCH 4
#define NTOK  (BATCH * SEQ)   // 8192
#define KC    32              // K-slice depth for GEMM staging

#if __has_builtin(__builtin_amdgcn_global_load_async_to_lds_b128)
#define HAVE_ASYNC_LDS 1
#endif

static __device__ inline uint16_t f2bf(float f) {
    uint32_t u = __float_as_uint(f);
    uint32_t r = (u + 0x7FFFu + ((u >> 16) & 1u)) >> 16;   // RNE
    return (uint16_t)r;
}

static __device__ inline v16bf as_bf(v16s v) { return __builtin_bit_cast(v16bf, v); }

// 16-byte global -> LDS copy. Async (gfx1250 GLOBAL_LOAD_ASYNC_TO_LDS_B128,
// tracked by ASYNCcnt) when the builtin exists; sync load+ds_write otherwise.
static __device__ inline void async_copy16(const uint16_t* g, uint16_t* l) {
#ifdef HAVE_ASYNC_LDS
    __builtin_amdgcn_global_load_async_to_lds_b128(
        (__attribute__((address_space(1))) v4i*)g,
        (__attribute__((address_space(3))) v4i*)l, 0, 0);
#else
    *(uint4*)l = *(const uint4*)g;
#endif
}

static __device__ inline void wait_async() {
#ifdef HAVE_ASYNC_LDS
#if __has_builtin(__builtin_amdgcn_s_wait_asynccnt)
    __builtin_amdgcn_s_wait_asynccnt(0);
#else
    asm volatile("s_wait_asynccnt 0x0" ::: "memory");
#endif
#endif
}

// A-operand fragment (16x32 bf16): lane holds row m = lane&15, half h = lane>>4;
// element pair (2v,2v+1) <-> k = 16h+2v+{0,1}. `row` = per-lane row
// pointer, contraction axis contiguous, 4B aligned.
static __device__ inline v16s load_afrag(const uint16_t* row, int h) {
    const uint32_t* p32 = (const uint32_t*)row;
    v16s r;
#pragma unroll
    for (int v = 0; v < 8; ++v) {
        int kbase = h * 16 + v * 2;
        uint32_t d = p32[kbase >> 1];
        r[2 * v]     = (short)(d & 0xFFFFu);
        r[2 * v + 1] = (short)(d >> 16);
    }
    return r;
}

// B-operand fragment (32x16 bf16): lane holds column n = lane&15; elements are
// 16 contiguous k starting at k0 = 16*(lane>>4).
static __device__ inline v16s load_bfrag(const uint16_t* col, int h) {
    const uint32_t* p32 = (const uint32_t*)(col + h * 16);
    v16s r;
#pragma unroll
    for (int v = 0; v < 8; ++v) {
        uint32_t d = p32[v];
        r[2 * v]     = (short)(d & 0xFFFFu);
        r[2 * v + 1] = (short)(d >> 16);
    }
    return r;
}

// ---------------- kernel 1: weight fp32 -> bf16 ----------------
__global__ void wcvt_kernel(const float* __restrict__ w, uint16_t* __restrict__ wb) {
    int i = (blockIdx.x * blockDim.x + threadIdx.x) * 4;   // covers 3*E*E exactly
    float4 v = *(const float4*)(w + i);
    wb[i + 0] = f2bf(v.x);
    wb[i + 1] = f2bf(v.y);
    wb[i + 2] = f2bf(v.z);
    wb[i + 3] = f2bf(v.w);
}

// ---------------- kernel 2: LayerNorm -> bf16 ----------------
__global__ void ln_kernel(const float* __restrict__ x, const float* __restrict__ g,
                          const float* __restrict__ bta, uint16_t* __restrict__ xb) {
    int token = blockIdx.x;
    int tid = threadIdx.x;
    const float* xr = x + (size_t)token * EMBED;
    float4 v = *(const float4*)(xr + tid * 4);
    float s  = v.x + v.y + v.z + v.w;
    float ss = v.x * v.x + v.y * v.y + v.z * v.z + v.w * v.w;
#pragma unroll
    for (int off = 16; off >= 1; off >>= 1) {
        s  += __shfl_xor(s, off);
        ss += __shfl_xor(ss, off);
    }
    __shared__ float rs[8], rss[8];
    int wid = tid >> 5, lane = tid & 31;
    if (lane == 0) { rs[wid] = s; rss[wid] = ss; }
    __syncthreads();
    if (tid == 0) {
        float ts = 0.f, tss = 0.f;
#pragma unroll
        for (int i = 0; i < 8; ++i) { ts += rs[i]; tss += rss[i]; }
        rs[0] = ts; rss[0] = tss;
    }
    __syncthreads();
    float mean = rs[0] * (1.0f / EMBED);
    float var  = rss[0] * (1.0f / EMBED) - mean * mean;
    float rstd = rsqrtf(var + 1e-5f);
    int base = tid * 4;
    float e0 = (v.x - mean) * rstd * g[base + 0] + bta[base + 0];
    float e1 = (v.y - mean) * rstd * g[base + 1] + bta[base + 1];
    float e2 = (v.z - mean) * rstd * g[base + 2] + bta[base + 2];
    float e3 = (v.w - mean) * rstd * g[base + 3] + bta[base + 3];
    uint16_t* o = xb + (size_t)token * EMBED + base;
    o[0] = f2bf(e0); o[1] = f2bf(e1); o[2] = f2bf(e2); o[3] = f2bf(e3);
}

// ---------------- kernel 3: QKV GEMM, LDS-tiled + async staging ----------------
// 256-thread block computes a 64(M) x 128(N) tile of y = x_hat @ W^T.
// Waves: wm = wave&3 picks 16-row strip, wn = wave>>2 picks 64-col strip.
// A 64xKC and B 128xKC slices are double-buffered in LDS, filled with
// GLOBAL_LOAD_ASYNC_TO_LDS_B128 and consumed via ds reads into WMMA fragments.
__global__ void qkv_gemm_kernel(const uint16_t* __restrict__ xb, const uint16_t* __restrict__ wb,
                                uint16_t* __restrict__ Qb, uint16_t* __restrict__ Kb,
                                uint16_t* __restrict__ Vt) {
    __shared__ __align__(16) uint16_t Abuf[2][64 * KC];    //  8 KB
    __shared__ __align__(16) uint16_t Bbuf[2][128 * KC];   // 16 KB
    int tid  = threadIdx.x;
    int wave = tid >> 5, lane = tid & 31;
    int n16 = lane & 15, hh = lane >> 4;
    int wm = wave & 3, wn = wave >> 2;

    int mt = blockIdx.x & 127;          // 128 M tiles of 64 rows
    int nt = blockIdx.x >> 7;           // 24  N tiles of 128 cols
    int s0 = mt * 64;
    int f0 = nt * 128;

    int arow = tid >> 2;                // 0..63
    int ac8  = (tid & 3) * 8;           // 16B column chunk
    const uint16_t* gA  = xb + (size_t)(s0 + arow) * EMBED + ac8;
    const uint16_t* gB0 = wb + (size_t)(f0 + arow) * EMBED + ac8;
    const uint16_t* gB1 = wb + (size_t)(f0 + 64 + arow) * EMBED + ac8;

    auto stage = [&](int kc, int buf) {
        int e0 = kc * KC;
        async_copy16(gA + e0,  &Abuf[buf][arow * KC + ac8]);
        async_copy16(gB0 + e0, &Bbuf[buf][arow * KC + ac8]);
        async_copy16(gB1 + e0, &Bbuf[buf][(64 + arow) * KC + ac8]);
    };

    v8f acc[4] = {};
    stage(0, 0);
    for (int kc = 0; kc < EMBED / KC; ++kc) {
        int cur = kc & 1;
        wait_async();            // my slice of buf[cur] landed
        __syncthreads();         // everyone's slice landed; prev reads of buf[cur^1] done
        if (kc + 1 < EMBED / KC) stage(kc + 1, cur ^ 1);
        v16s a = load_afrag(&Abuf[cur][(wm * 16 + n16) * KC], hh);
#pragma unroll
        for (int t = 0; t < 4; ++t) {
            v16s b = load_bfrag(&Bbuf[cur][(wn * 64 + t * 16 + n16) * KC], hh);
            acc[t] = __builtin_amdgcn_wmma_f32_16x16x32_bf16(
                false, as_bf(a), false, as_bf(b), (short)0, acc[t], false, false);
        }
    }

    // Epilogue: scatter into Q[B,H,S,D] (scaled), K[B,H,S,D], Vt[B,H,D,S].
    int fbase  = f0 + wn * 64;                                        // wave-uniform, 64-aligned
    int region = __builtin_amdgcn_readfirstlane(fbase >> 10);         // 0=Q 1=K 2=V
    int head   = __builtin_amdgcn_readfirstlane((fbase >> 6) & 15);
    int srow0  = s0 + wm * 16;
#pragma unroll
    for (int t = 0; t < 4; ++t) {
        int d = t * 16 + n16;
#pragma unroll
        for (int r = 0; r < 8; ++r) {
            int s = srow0 + r + 8 * hh;
            int bi = s >> 11, seq = s & 2047;
            float v = acc[t][r];
            if (region == 0) {
                Qb[((size_t)(bi * NHEAD + head) * SEQ + seq) * HDIM + d] = f2bf(v * 0.125f);
            } else if (region == 1) {
                Kb[((size_t)(bi * NHEAD + head) * SEQ + seq) * HDIM + d] = f2bf(v);
            } else {
                Vt[((size_t)(bi * NHEAD + head) * HDIM + d) * SEQ + seq] = f2bf(v);
            }
        }
    }
}

// ---------------- kernel 4: causal flash attention ----------------
// One wave per (b, h, 16-query tile). Online softmax, 32-key blocks.
__global__ void attn_kernel(const uint16_t* __restrict__ Qb, const uint16_t* __restrict__ Kb,
                            const uint16_t* __restrict__ Vt, const float* __restrict__ resid,
                            float* __restrict__ out) {
    __shared__ __align__(16) uint16_t pbuf[8 * 16 * 32];   // per-wave P staging (C->A transpose)
    int wave = threadIdx.x >> 5;
    int lane = threadIdx.x & 31;
    int wg = blockIdx.x * 8 + wave;           // 8192 waves total
    int qt = wg & 127;
    int h  = (wg >> 7) & 15;
    int b  = wg >> 11;
    int q0 = qt * 16;
    int n16 = lane & 15, hh = lane >> 4;

    const uint16_t* Qbase  = Qb + (size_t)(b * NHEAD + h) * SEQ * HDIM;
    const uint16_t* Kbase  = Kb + (size_t)(b * NHEAD + h) * SEQ * HDIM;
    const uint16_t* Vtbase = Vt + (size_t)(b * NHEAD + h) * HDIM * SEQ;

    const uint16_t* qrow = Qbase + (size_t)(q0 + n16) * HDIM;
    v16s qa0 = load_afrag(qrow, hh);
    v16s qa1 = load_afrag(qrow + 32, hh);

    v8f acc[4] = {};
    float m_run[8], l_run[8];
#pragma unroll
    for (int r = 0; r < 8; ++r) { m_run[r] = -1e30f; l_run[r] = 0.0f; }

    uint16_t* pw = pbuf + wave * 512;

    for (int j = 0; j <= q0 + 15; j += 32) {
        // prefetch next key/value block (one cacheline per lane covers the tile)
        if (j + 32 <= q0 + 15) {
            __builtin_prefetch(Kbase + (size_t)(j + 32 + lane) * HDIM, 0, 1);
            __builtin_prefetch(Vtbase + (size_t)(lane * 2) * SEQ + (j + 32), 0, 1);
        }
        // ---- scores: S = Q(16x64) . K^T(64x32) -> two 16x16 f32 tiles
        v8f s[2];
#pragma unroll
        for (int t = 0; t < 2; ++t) {
            const uint16_t* kcol = Kbase + (size_t)(j + t * 16 + n16) * HDIM;  // B[k=d][n=key]
            v16s b_lo = load_bfrag(kcol, hh);
            v16s b_hi = load_bfrag(kcol + 32, hh);
            v8f z = {};
            z = __builtin_amdgcn_wmma_f32_16x16x32_bf16(false, as_bf(qa0), false, as_bf(b_lo),
                                                        (short)0, z, false, false);
            z = __builtin_amdgcn_wmma_f32_16x16x32_bf16(false, as_bf(qa1), false, as_bf(b_hi),
                                                        (short)0, z, false, false);
            s[t] = z;
        }
        // ---- causal mask (finite -inf keeps exp well-defined)
#pragma unroll
        for (int t = 0; t < 2; ++t) {
            if (j + t * 16 + 15 > q0) {
                int key = j + t * 16 + n16;
#pragma unroll
                for (int r = 0; r < 8; ++r) {
                    int qrowi = q0 + r + 8 * hh;
                    if (key > qrowi) s[t][r] = -1e30f;
                }
            }
        }
        // ---- online softmax per row (rows per-register; reduce across 16-lane half)
        float alpha[8];
#pragma unroll
        for (int r = 0; r < 8; ++r) {
            float pm = fmaxf(s[0][r], s[1][r]);
            pm = fmaxf(pm, __shfl_xor(pm, 1));
            pm = fmaxf(pm, __shfl_xor(pm, 2));
            pm = fmaxf(pm, __shfl_xor(pm, 4));
            pm = fmaxf(pm, __shfl_xor(pm, 8));
            float mn = fmaxf(m_run[r], pm);
            alpha[r] = __expf(m_run[r] - mn);
            m_run[r] = mn;
            float p0 = __expf(s[0][r] - mn);
            float p1 = __expf(s[1][r] - mn);
            s[0][r] = p0; s[1][r] = p1;
            float ps = p0 + p1;
            ps += __shfl_xor(ps, 1);
            ps += __shfl_xor(ps, 2);
            ps += __shfl_xor(ps, 4);
            ps += __shfl_xor(ps, 8);
            l_run[r] = l_run[r] * alpha[r] + ps;
        }
#pragma unroll
        for (int t = 0; t < 4; ++t)
#pragma unroll
            for (int r = 0; r < 8; ++r) acc[t][r] *= alpha[r];

        // ---- P: C-layout -> A-layout via LDS (bf16)
#pragma unroll
        for (int t = 0; t < 2; ++t)
#pragma unroll
            for (int r = 0; r < 8; ++r)
                pw[(r + 8 * hh) * 32 + t * 16 + n16] = f2bf(s[t][r]);
        __builtin_amdgcn_wave_barrier();
        v16s pa = load_afrag(pw + n16 * 32, hh);
        __builtin_amdgcn_wave_barrier();

        // ---- O += P(16x32) . V(32x64); Vt is [d][s]: B columns contiguous in key
#pragma unroll
        for (int t = 0; t < 4; ++t) {
            const uint16_t* vcol = Vtbase + (size_t)(t * 16 + n16) * SEQ + j;
            v16s vb = load_bfrag(vcol, hh);
            acc[t] = __builtin_amdgcn_wmma_f32_16x16x32_bf16(
                false, as_bf(pa), false, as_bf(vb), (short)0, acc[t], false, false);
        }
    }

    // ---- epilogue: normalize, add residual
    float inv[8];
#pragma unroll
    for (int r = 0; r < 8; ++r) inv[r] = 1.0f / l_run[r];
#pragma unroll
    for (int t = 0; t < 4; ++t) {
        int e = h * HDIM + t * 16 + n16;
#pragma unroll
        for (int r = 0; r < 8; ++r) {
            int seq = q0 + r + 8 * hh;
            size_t idx = ((size_t)(b * SEQ + seq)) * EMBED + e;
            out[idx] = acc[t][r] * inv[r] + resid[idx];
        }
    }
}

// ---------------- launcher ----------------
extern "C" void kernel_launch(void* const* d_in, const int* in_sizes, int n_in,
                              void* d_out, int out_size, void* d_ws, size_t ws_size,
                              hipStream_t stream) {
    (void)in_sizes; (void)n_in; (void)out_size; (void)ws_size;
    const float* inputs = (const float*)d_in[0];
    const float* ln_w   = (const float*)d_in[1];
    const float* ln_b   = (const float*)d_in[2];
    const float* w_qkv  = (const float*)d_in[3];
    float* out = (float*)d_out;

    char* ws = (char*)d_ws;
    uint16_t* xb = (uint16_t*)(ws);                               // 16 MB  (8192*1024 bf16)
    uint16_t* wb = (uint16_t*)(ws + (size_t)16 * 1024 * 1024);    //  6 MB  (3072*1024 bf16)
    uint16_t* Qb = (uint16_t*)(ws + (size_t)22 * 1024 * 1024);    // 16 MB
    uint16_t* Kb = (uint16_t*)(ws + (size_t)38 * 1024 * 1024);    // 16 MB
    uint16_t* Vt = (uint16_t*)(ws + (size_t)54 * 1024 * 1024);    // 16 MB -> 70 MB total

    wcvt_kernel<<<3072, 256, 0, stream>>>(w_qkv, wb);              // 3072*256*4 = 3*E*E
    ln_kernel<<<NTOK, 256, 0, stream>>>(inputs, ln_w, ln_b, xb);
    qkv_gemm_kernel<<<3072, 256, 0, stream>>>(xb, wb, Qb, Kb, Vt); // 128 x 24 tiles
    attn_kernel<<<1024, 256, 0, stream>>>(Qb, Kb, Vt, inputs, out); // 8192 waves
}
